// CAModel_9783935500964
// MI455X (gfx1250) — matrix operand order repaired and checked by reading, not compiled, not run on hardware
//
#include <hip/hip_runtime.h>
#include <hip/hip_bf16.h>
#include <cstdint>

// NCA: 4 steps of perceive(sobel) + per-pixel MLP(48->128->16) + fire mask + life gate.
// Per-pixel MLP mapped to v_wmma_f32_16x16x32_f16 (N=16 pixels per WMMA tile).
// Bias folded into the GEMM via K=48 slot (y[48]=1, w0[:,48]=fc0_b).
// 20 WMMAs per 16 pixels; all inter-step traffic L2-resident (x is 8MB, L2 192MB).

typedef __attribute__((ext_vector_type(16))) _Float16 v16h;
typedef __attribute__((ext_vector_type(8)))  float    v8f;

#define CH     16
#define HID    128
#define HH     256
#define WW     256
#define NB     8
#define NSTEPS 4

// LDS layout (bytes)
#define XS_OFF  0                    // float xs[16][18][19]
#define XS_BYTES (16*18*19*4)        // 21888
#define W0_OFF  (XS_OFF + XS_BYTES)  // _Float16 w0[128][64]  (k<48: w, k==48: bias, else 0)
#define W0_BYTES (128*64*2)
#define W1_OFF  (W0_OFF + W0_BYTES)  // _Float16 w1[16][128]
#define W1_BYTES (16*128*2)
#define Y_OFF   (W1_OFF + W1_BYTES)  // _Float16 ybuf[256][136] (y then reused for h)
#define Y_STRIDE 136
#define Y_BYTES (256*Y_STRIDE*2)
#define SMEM_BYTES (Y_OFF + Y_BYTES) // 112000

__device__ __forceinline__ v16h ld16h(const _Float16* p0, const _Float16* p1) {
  union { v16h v; uint4 q[2]; } u;
  u.q[0] = *(const uint4*)p0;
  u.q[1] = *(const uint4*)p1;
  return u.v;
}

__global__ __launch_bounds__(256) void nca_step_kernel(
    const float* __restrict__ xin,    // [8,16,256,256]
    const float* __restrict__ fc0_w,  // [128,48]
    const float* __restrict__ fc0_b,  // [128]
    const float* __restrict__ fc1_w,  // [16,128]
    const float* __restrict__ maskp,  // [8,256,256] (this step's slice)
    float*       __restrict__ xn,     // [8,16,256,256]
    unsigned char* __restrict__ pre)  // [8,256,256]
{
  extern __shared__ char smem[];
  float*    xs   = (float*)(smem + XS_OFF);
  _Float16* w0   = (_Float16*)(smem + W0_OFF);
  _Float16* w1   = (_Float16*)(smem + W1_OFF);
  _Float16* ybuf = (_Float16*)(smem + Y_OFF);

  const int tid = threadIdx.x;
  const int x0 = blockIdx.x * 16;
  const int y0 = blockIdx.y * 16;
  const int b  = blockIdx.z;

  // ---- weights -> LDS (f16; bias folded into K=48 column; 49..63 zero) ----
  for (int i = tid; i < 128 * 64; i += 256) {
    int m = i >> 6, k = i & 63;
    float v = (k < 48) ? fc0_w[m * 48 + k] : ((k == 48) ? fc0_b[m] : 0.0f);
    w0[i] = (_Float16)v;
  }
  for (int i = tid; i < 16 * 128; i += 256) w1[i] = (_Float16)fc1_w[i];

  // ---- x tile + halo -> LDS (zero pad OOB; safe for conv and for >0.1 maxpool) ----
  const float* xb = xin + (size_t)b * CH * HH * WW;
  for (int i = tid; i < 16 * 18 * 18; i += 256) {
    int c = i / (18 * 18);
    int r = i % (18 * 18);
    int yy = r / 18, xx = r % 18;
    int gy = y0 + yy - 1, gx = x0 + xx - 1;
    float v = 0.0f;
    if (gy >= 0 && gy < HH && gx >= 0 && gx < WW) v = xb[c * (HH*WW) + gy * WW + gx];
    xs[(c * 18 + yy) * 19 + xx] = v;
  }
  __syncthreads();

  // ---- perceive: y[48] per pixel (+1.0 bias slot) -> f16 LDS; plus pre-life mask ----
  {
    const int py = tid >> 4, px = tid & 15;
    _Float16* yrow = ybuf + tid * Y_STRIDE;
#pragma unroll
    for (int c = 0; c < CH; ++c) {
      const float* p = xs + (c * 18 + py) * 19 + px;  // 3x3 top-left (halo coords)
      float a00=p[0],  a01=p[1],  a02=p[2];
      float a10=p[19], a11=p[20], a12=p[21];
      float a20=p[38], a21=p[39], a22=p[40];
      float sx = ((a02 - a00) + 2.0f*(a12 - a10) + (a22 - a20)) * 0.125f;
      float sy = ((a20 - a00) + 2.0f*(a21 - a01) + (a22 - a02)) * 0.125f;
      yrow[c]        = (_Float16)a11;
      yrow[16 + c]   = (_Float16)sx;
      yrow[32 + c]   = (_Float16)sy;
    }
    yrow[48] = (_Float16)1.0f;           // bias multiplier
#pragma unroll
    for (int k = 49; k < 64; ++k) yrow[k] = (_Float16)0.0f;

    const float* p3 = xs + (3 * 18 + py) * 19 + px;
    float mm = fmaxf(fmaxf(fmaxf(p3[0],p3[1]),p3[2]),
               fmaxf(fmaxf(fmaxf(p3[19],p3[20]),p3[21]),
                     fmaxf(fmaxf(p3[38],p3[39]),p3[40])));
    pre[((size_t)b * HH + (y0 + py)) * WW + (x0 + px)] = (mm > 0.1f) ? 1 : 0;
  }
  __syncthreads();

  // ---- per-wave GEMMs over two 16-pixel groups ----
  const int lane = tid & 31;
  const int wave = tid >> 5;
  const int lo = lane & 15;   // column / row-in-tile
  const int hi = lane >> 4;   // half-wave selector

  float* xnb = xn + (size_t)b * CH * HH * WW;

  for (int pgi = 0; pgi < 2; ++pgi) {
    const int pg = wave * 2 + pgi;            // pixel group 0..15
    _Float16* yr = ybuf + (pg * 16 + lo) * Y_STRIDE;

    // B fragments of GEMM1 (y: 49x16, K padded to 64)
    v16h bf0 = ld16h(yr + 8 * hi,      yr + 16 + 8 * hi);        // K = 0..31
    v16h bf1 = ld16h(yr + 32 + 8 * hi, yr + 48 + 8 * hi);   // K = 32..63

    // GEMM1: h[128x16] = [fc0_w | fc0_b] x [y ; 1]  -- batch A loads, then WMMA chain
    v16h afr[8];
    v8f  acc[8];
#pragma unroll
    for (int mt = 0; mt < 8; ++mt) {
      const _Float16* w = w0 + (mt * 16 + lo) * 64;
      afr[mt] = ld16h(w + 8 * hi, w + 16 + 8 * hi);               // K-tile 0
    }
#pragma unroll
    for (int mt = 0; mt < 8; ++mt) {
      v8f c = {};
      acc[mt] = __builtin_amdgcn_wmma_f32_16x16x32_f16(false, afr[mt], false, bf0,
                                                       (short)0, c, false, false);
    }
#pragma unroll
    for (int mt = 0; mt < 8; ++mt) {
      const _Float16* w = w0 + (mt * 16 + lo) * 64;
      afr[mt] = ld16h(w + 32 + 8 * hi, w + 48 + 8 * hi);          // K-tile 1
    }
#pragma unroll
    for (int mt = 0; mt < 8; ++mt) {
      acc[mt] = __builtin_amdgcn_wmma_f32_16x16x32_f16(false, afr[mt], false, bf1,
                                                       (short)0, acc[mt], false, false);
    }

    // relu -> f16, write h into same same-wave LDS rows (in-order DS, no barrier)
#pragma unroll
    for (int mt = 0; mt < 8; ++mt) {
      union { uint4 q; _Float16 h[8]; } u;
#pragma unroll
      for (int e = 0; e < 8; ++e)
        u.h[e] = (_Float16)fmaxf(acc[mt][e], 0.0f);
      *(uint4*)(yr + mt * 16 + 8 * hi) = u.q;   // k = mt*16 + 8*hi .. +7
    }

    // GEMM2: delta[16x16] = fc1_w x h  -- batch all fragment loads, then 4 WMMAs
    v16h a1[4], bh[4];
#pragma unroll
    for (int kt = 0; kt < 4; ++kt) {
      const _Float16* src = w1 + lo * 128 + kt * 32 + 8 * hi;
      a1[kt] = ld16h(src, src + 16);
    }
#pragma unroll
    for (int kt = 0; kt < 4; ++kt)
      bh[kt] = ld16h(yr + kt * 32 + 8 * hi, yr + kt * 32 + 16 + 8 * hi);
    v8f d = {};
#pragma unroll
    for (int kt = 0; kt < 4; ++kt)
      d = __builtin_amdgcn_wmma_f32_16x16x32_f16(false, a1[kt], false, bh[kt],
                                                 (short)0, d, false, false);

    // epilogue: xn = x + delta * mask  (lane holds pixel pg*16+lo, channels e+8*hi)
    const int pix = pg * 16 + lo;
    const int py = pix >> 4, px = pix & 15;
    const int gy = y0 + py, gx = x0 + px;
    const float mk = maskp[((size_t)b * HH + gy) * WW + gx];
#pragma unroll
    for (int e = 0; e < 8; ++e) {
      int c = e + 8 * hi;
      float xv = xs[(c * 18 + py + 1) * 19 + (px + 1)];
      xnb[c * (HH*WW) + gy * WW + gx] = xv + d[e] * mk;
    }
  }
}

__global__ __launch_bounds__(256) void nca_life_kernel(
    const float* __restrict__ xnbuf,
    const unsigned char* __restrict__ pre,
    float* __restrict__ xout)
{
  const int idx = blockIdx.x * 256 + threadIdx.x;   // over B*H*W
  const int gx = idx & 255;
  const int gy = (idx >> 8) & 255;
  const int b  = idx >> 16;
  const float* xb = xnbuf + (size_t)b * CH * HH * WW;

  float mm = -1e30f;
#pragma unroll
  for (int dy = -1; dy <= 1; ++dy) {
    int yy = gy + dy;
    if (yy < 0 || yy >= HH) continue;
#pragma unroll
    for (int dx = -1; dx <= 1; ++dx) {
      int xx = gx + dx;
      if (xx < 0 || xx >= WW) continue;
      mm = fmaxf(mm, xb[3 * (HH*WW) + yy * WW + xx]);
    }
  }
  const float life = ((mm > 0.1f) && (pre[idx] != 0)) ? 1.0f : 0.0f;
  float* ob = xout + (size_t)b * CH * HH * WW;
#pragma unroll
  for (int c = 0; c < CH; ++c)
    ob[c * (HH*WW) + gy * WW + gx] = xb[c * (HH*WW) + gy * WW + gx] * life;
}

extern "C" void kernel_launch(void* const* d_in, const int* in_sizes, int n_in,
                              void* d_out, int out_size, void* d_ws, size_t ws_size,
                              hipStream_t stream) {
  const float* x      = (const float*)d_in[0];
  const float* fc0_w  = (const float*)d_in[1];
  const float* fc0_b  = (const float*)d_in[2];
  const float* fc1_w  = (const float*)d_in[3];
  const float* mask   = (const float*)d_in[4];
  float* out = (float*)d_out;

  const size_t NELEM = (size_t)NB * CH * HH * WW;   // 2,097,152
  char* ws = (char*)d_ws;
  float*         xnbuf = (float*)ws;                         // 8 MB
  float*         ping  = (float*)(ws + NELEM * 4);           // 8 MB
  unsigned char* pre   = (unsigned char*)(ws + NELEM * 8);   // 512 KB

  // allow >64KB dynamic LDS (gfx1250 WGP has 320KB)
  hipFuncSetAttribute((const void*)nca_step_kernel,
                      hipFuncAttributeMaxDynamicSharedMemorySize, SMEM_BYTES);

  dim3 g1(WW / 16, HH / 16, NB), b1(256);
  dim3 g2((unsigned)(NELEM / CH / 256)), b2(256);

  const float* cur = x;
  float* outs[NSTEPS] = { ping, out, ping, out };
  for (int s = 0; s < NSTEPS; ++s) {
    const float* mk = mask + (size_t)s * NB * HH * WW;
    nca_step_kernel<<<g1, b1, SMEM_BYTES, stream>>>(cur, fc0_w, fc0_b, fc1_w, mk, xnbuf, pre);
    nca_life_kernel<<<g2, b2, 0, stream>>>(xnbuf, pre, outs[s]);
    cur = outs[s];
  }
}
